// AdditiveAttention_10574209483482
// MI455X (gfx1250) — matrix operand order repaired and checked by reading, not compiled, not run on hardware
//
#include <hip/hip_runtime.h>
#include <hip/hip_bf16.h>

// Additive attention for MI455X (gfx1250, wave32).
// Score stage: trans-pipe bound (v_tanh_f32). PV stage: v_wmma_f32_16x16x32_bf16.
// Softmax stats are computed in the A-layout lane mapping so that row reductions
// are in-lane (one shfl_xor(16) per reduction) instead of ds_bpermute trees.

typedef __attribute__((ext_vector_type(16))) __bf16 v16bf;
typedef __attribute__((ext_vector_type(8)))  float  v8f;

#define B_ 2
#define L_ 512
#define S_ 512
#define H_ 8
#define E_ 32
#define D_ 64
#define STILE 32

__device__ __forceinline__ float fast_tanh(float x) {
#if __has_builtin(__builtin_amdgcn_tanhf)
    return __builtin_amdgcn_tanhf(x);          // v_tanh_f32 (CDNA5 trans op)
#elif __has_builtin(__builtin_amdgcn_tanh_f32)
    return __builtin_amdgcn_tanh_f32(x);
#else
    float e = __expf(2.0f * x);                // v_exp_f32 fallback
    return 1.0f - 2.0f / (e + 1.0f);
#endif
}

__global__ __launch_bounds__(32) void AdditiveAttention_10574209483482_kernel(
    const float* __restrict__ Q,     // (B,L,H,E)
    const float* __restrict__ K,     // (B,S,H,E)
    const float* __restrict__ V,     // (B,S,H,D)
    const float* __restrict__ vvec,  // (E)
    const float* __restrict__ MASK,  // (L,S)
    const float* __restrict__ KLEN,  // (B,S)
    float* __restrict__ OUT)         // (B,L,H,D)
{
    __shared__ __align__(16) float sc_lds[16 * STILE]; // 2 KB fp32 score tile
    __shared__ __align__(16) float bc_lds[32];         // per-row broadcast buffer

    const int lane = threadIdx.x & 31;
    const int bid  = blockIdx.x;
    const int lt   = bid & (L_ / 16 - 1);     // 32 L-tiles
    const int h    = (bid >> 5) & (H_ - 1);
    const int b    = bid >> 8;
    const int l0   = lt * 16;

    // A/C layout lane mapping: this lane owns row `arow`; its K-set starts at
    // kbase: lanes 0-15 -> K {0..7,16..23}, lanes 16-31 -> K {8..15,24..31}.
    // kbase also equals the C-layout M base (VGPR j <-> M = kbase + j ... j+8 split).
    const int arow  = lane & 15;
    const int kbase = (lane < 16) ? 0 : 8;

    // v vector: uniform -> scalar regs.
    float vv[E_];
#pragma unroll
    for (int e = 0; e < E_; ++e) vv[e] = vvec[e];

    // Online-softmax running state for row `arow` (duplicated in lane & lane+16).
    float m_run = -1e30f;
    float l_run = 0.0f;

    v8f acc[4];
#pragma unroll
    for (int n = 0; n < 4; ++n)
#pragma unroll
        for (int j = 0; j < 8; ++j) acc[n][j] = 0.0f;

    const int nN = lane & 15;                 // B layout: column within n-tile
    const int kb = (lane < 16) ? 0 : 16;      // B layout: K half

    for (int st = 0; st < S_ / STILE; ++st) {
        const int s0 = st * STILE;
        const int s  = s0 + lane;             // this lane's score column

        // Cache this lane's K row in VGPRs (contiguous 128B -> b128 clauses, L2-served).
        const float* krow = K + (((b * S_ + s) * H_ + h) * E_);
        float kk[E_];
#pragma unroll
        for (int e = 0; e < E_; ++e) kk[e] = krow[e];
        const float kl = KLEN[b * S_ + s];

        // ---- score stage: 16 rows x 32 cols x 32 E (v_tanh_f32 bound) ----
        float p[16];
#pragma unroll 4
        for (int r = 0; r < 16; ++r) {
            const float* qrow = Q + (((b * L_ + (l0 + r)) * H_ + h) * E_);
            float sc = 0.0f;
#pragma unroll
            for (int e = 0; e < E_; ++e)
                sc = fmaf(vv[e], fast_tanh(qrow[e] + kk[e]), sc);
            sc += MASK[(l0 + r) * S_ + s] + kl;
            p[r] = sc;
        }

        // ---- transpose fp32 scores into A-layout lane mapping via LDS ----
#pragma unroll
        for (int r = 0; r < 16; ++r)
            sc_lds[r * STILE + lane] = p[r];
        __syncthreads();

        float f[16];  // this lane's 16 A-layout K values of row arow (fp32)
#pragma unroll
        for (int j = 0; j < 8; ++j) f[j]     = sc_lds[arow * STILE + kbase + j];
#pragma unroll
        for (int j = 0; j < 8; ++j) f[8 + j] = sc_lds[arow * STILE + kbase + 16 + j];

        // ---- row stats: in-lane reduce + one cross-half shuffle ----
        float mx = f[0];
#pragma unroll
        for (int j = 1; j < 16; ++j) mx = fmaxf(mx, f[j]);
        mx = fmaxf(mx, __shfl_xor(mx, 16, 32));     // combine K halves of the row

        const float mn    = fmaxf(m_run, mx);
        const float alpha = __expf(m_run - mn);
        m_run = mn;

        float sum = 0.0f;
        v16bf A;
#pragma unroll
        for (int j = 0; j < 16; ++j) {
            const float pe = __expf(f[j] - mn);
            sum += pe;
            A[j] = (__bf16)pe;                      // A operand assembled in-lane
        }
        sum += __shfl_xor(sum, 16, 32);
        l_run = l_run * alpha + sum;

        // ---- broadcast alpha to C-layout rows (2 contiguous LDS reads) ----
        bc_lds[lane] = alpha;                       // rows 0..15 live at idx 0..15
        __syncthreads();
        float al[8];
#pragma unroll
        for (int j = 0; j < 8; ++j) al[j] = bc_lds[kbase + j];

#pragma unroll
        for (int n = 0; n < 4; ++n)
#pragma unroll
            for (int j = 0; j < 8; ++j)
                acc[n][j] *= al[j];

        // ---- P(16x32) x V(32x16) per n-tile, D=64 -> 4 WMMAs ----
#pragma unroll
        for (int n = 0; n < 4; ++n) {
            // B layout (16-bit, 32x16): lane = column N (mod 16), element j holds
            // K = kb/2 + j (j<8), 16 + kb/2 + (j-8) (j>=8). Loads are d-coalesced (64B runs).
            v16bf Bm;
#pragma unroll
            for (int j = 0; j < 16; ++j) {
                const float fv = V[((b * S_ + (s0 + (kb >> 1) + j + (j & 8))) * H_ + h) * D_ + n * 16 + nN];
                Bm[j] = (__bf16)fv;
            }
            acc[n] = __builtin_amdgcn_wmma_f32_16x16x32_bf16(
                false, A, false, Bm, (short)0, acc[n], false, false);
        }
        __syncthreads();   // sc_lds/bc_lds reused next iteration
    }

    // ---- normalize by row sums and store (B,L,H,D) ----
    bc_lds[lane] = 1.0f / l_run;                   // inv for row arow
    __syncthreads();
    float inv[8];
#pragma unroll
    for (int j = 0; j < 8; ++j) inv[j] = bc_lds[kbase + j];

#pragma unroll
    for (int n = 0; n < 4; ++n)
#pragma unroll
        for (int j = 0; j < 8; ++j) {
            const int M   = j + kbase;              // C layout: VGPR j <-> M = j / j+8
            const float o = acc[n][j] * inv[j];
            OUT[((b * L_ + (l0 + M)) * H_ + h) * D_ + n * 16 + nN] = o;
        }
}

extern "C" void kernel_launch(void* const* d_in, const int* in_sizes, int n_in,
                              void* d_out, int out_size, void* d_ws, size_t ws_size,
                              hipStream_t stream) {
    (void)in_sizes; (void)n_in; (void)d_ws; (void)ws_size; (void)out_size;
    const float* Q    = (const float*)d_in[0];  // queries (B,L,H,E)
    const float* K    = (const float*)d_in[1];  // keys    (B,S,H,E)
    const float* V    = (const float*)d_in[2];  // values  (B,S,H,D)
    const float* vvec = (const float*)d_in[3];  // v       (E)
    const float* MASK = (const float*)d_in[4];  // attn_mask_add (L,S)
    const float* KLEN = (const float*)d_in[5];  // key_lengths_add (B,S)
    float* OUT = (float*)d_out;                 // (B,L,H,D)

    const int nblocks = B_ * H_ * (L_ / 16);    // 512 wave32 workgroups
    AdditiveAttention_10574209483482_kernel<<<nblocks, 32, 0, stream>>>(
        Q, K, V, vvec, MASK, KLEN, OUT);
}
